// GPT_33973191311506
// MI455X (gfx1250) — matrix operand order repaired and checked by reading, not compiled, not run on hardware
//
#include <hip/hip_runtime.h>
#include <hip/hip_bf16.h>
#include <stdint.h>

// ---------------------------------------------------------------------------
// GPT-2 (L=4, H=12, C=768, B=2, T=2048, V=50257) forward on gfx1250.
// GEMMs: v_wmma_f32_16x16x32_f16, software-pipelined register double-buffered
// global loads, 16x32 tile per wave.
// Attention: flash-style, per-wave async double-buffered K/V tiles in LDS
// (global_load_async_to_lds_b128 + s_wait_asynccnt, no barriers needed).
// ---------------------------------------------------------------------------

typedef _Float16 f16;
typedef __attribute__((ext_vector_type(16))) _Float16 v16h;
typedef __attribute__((ext_vector_type(8)))  _Float16 v8h;
typedef __attribute__((ext_vector_type(8)))  float    v8f;

union V16H { v16h v; v8h h[2]; };

#define GPT_B 2
#define GPT_T 2048
#define GPT_C 768
#define GPT_H 12
#define GPT_HD 64
#define GPT_L 4
#define GPT_V 50257
#define GPT_VPAD 50304      // round up to 128
#define GPT_M (GPT_B * GPT_T)
#define GPT_TP (GPT_T + 32) // padded key dim for V^T rows

// ---------------------------------------------------------------------------
// small helpers
// ---------------------------------------------------------------------------
__device__ __forceinline__ float blockReduce256(float v, float* sm, bool isMax) {
  // blockDim.x == 256 (8 waves of 32)
  #pragma unroll
  for (int off = 16; off; off >>= 1) {
    float o = __shfl_xor(v, off, 32);
    v = isMax ? fmaxf(v, o) : (v + o);
  }
  if ((threadIdx.x & 31) == 0) sm[threadIdx.x >> 5] = v;
  __syncthreads();
  if (threadIdx.x == 0) {
    float r = sm[0];
    #pragma unroll
    for (int i = 1; i < 8; ++i) r = isMax ? fmaxf(r, sm[i]) : (r + sm[i]);
    sm[0] = r;
  }
  __syncthreads();
  float out = sm[0];
  __syncthreads();
  return out;
}

// ---------------------------------------------------------------------------
// 1) token + position embedding:  x[m,c] = wte[idx[m],c] + wpe[m%T,c]
// ---------------------------------------------------------------------------
__global__ void embed_kernel(const int* __restrict__ idx,
                             const float* __restrict__ wte,
                             const float* __restrict__ wpe,
                             float* __restrict__ x) {
  size_t i = (size_t)blockIdx.x * blockDim.x + threadIdx.x;
  if (i >= (size_t)GPT_M * GPT_C) return;
  int m = (int)(i / GPT_C);
  int c = (int)(i % GPT_C);
  int t = m % GPT_T;
  x[i] = wte[(size_t)idx[m] * GPT_C + c] + wpe[(size_t)t * GPT_C + c];
}

// ---------------------------------------------------------------------------
// 2) LayerNorm (no bias) -> f16 output. One block (256 thr) per row, C=768.
// ---------------------------------------------------------------------------
__global__ __launch_bounds__(256) void ln_f16_kernel(const float* __restrict__ x,
                                                     const float* __restrict__ w,
                                                     f16* __restrict__ out) {
  __shared__ float sm[8];
  int row = blockIdx.x;
  const float* xr = x + (size_t)row * GPT_C;
  float s = 0.f, s2 = 0.f;
  for (int c = threadIdx.x; c < GPT_C; c += 256) {
    float v = xr[c]; s += v; s2 += v * v;
  }
  float sum  = blockReduce256(s,  sm, false);
  float sum2 = blockReduce256(s2, sm, false);
  float mean = sum * (1.0f / GPT_C);
  float var  = sum2 * (1.0f / GPT_C) - mean * mean;
  float rstd = rsqrtf(var + 1e-5f);
  for (int c = threadIdx.x; c < GPT_C; c += 256)
    out[(size_t)row * GPT_C + c] = (f16)((xr[c] - mean) * rstd * w[c]);
}

// ---------------------------------------------------------------------------
// 3) weight convert + transpose: W[K,N] f32 -> Wt[Npad,K] f16 (zero pad rows)
// ---------------------------------------------------------------------------
__global__ void convw_kernel(const float* __restrict__ W, f16* __restrict__ Wt,
                             int K, int N, int Npad) {
  size_t i = (size_t)blockIdx.x * 256 + threadIdx.x;
  if (i >= (size_t)Npad * K) return;
  int n = (int)(i / K);
  int k = (int)(i % K);
  float v = (n < N) ? W[(size_t)k * N + n] : 0.0f;
  Wt[i] = (f16)v;
}

// ---------------------------------------------------------------------------
// 4) WMMA GEMM:  Cout[M,ldc] (+= Cin) = A_f16[M,K] @ Wt_f16[Npad,K]^T
//    Block: 256 thr = 8 waves -> block tile 32(M) x 128(N); wave -> 16x32.
//    Register double-buffering: iteration i+1's fragments are loaded before
//    iteration i's WMMAs consume theirs, keeping >=6 b128 loads in flight
//    across each WMMA pair (partial s_wait_loadcnt instead of full drains).
// ---------------------------------------------------------------------------
__global__ __launch_bounds__(256) void gemm_f16_kernel(const f16* __restrict__ A,
                                                       const f16* __restrict__ Bt,
                                                       const float* __restrict__ Cin,
                                                       float* __restrict__ Cout,
                                                       int M, int Npad, int K,
                                                       int Nstore, int ldc) {
  const int tid  = threadIdx.x;
  const int lane = tid & 31;
  const int wid  = tid >> 5;
  const int half = lane >> 4;
  const int nl   = lane & 15;
  const int wm   = wid >> 2;                  // 0..1  (M direction)
  const int wn   = wid & 3;                   // 0..3  (N direction)
  const int tileM = blockIdx.y * 32 + wm * 16;
  const int tileN = blockIdx.x * 128 + wn * 32;

  const f16* arow = A  + (size_t)(tileM + nl) * K;
  const f16* b0   = Bt + (size_t)(tileN + nl) * K;
  const f16* b1   = Bt + (size_t)(tileN + 16 + nl) * K;

  v8f acc0 = {}, acc1 = {};

  // prologue: fragments for k0 = 0
  V16H a, bb0, bb1;
  a.v    = *(const v16h*)(arow + half * 16);
  bb0.v  = *(const v16h*)(b0 + half * 16);
  bb1.v  = *(const v16h*)(b1 + half * 16);

  for (int k0 = 0; k0 < K; k0 += 32) {
    // issue next iteration's loads first (last iter: redundant reload of k0)
    int kn = (k0 + 32 < K) ? k0 + 32 : k0;
    __builtin_prefetch(arow + kn + 128, 0, 1);
    __builtin_prefetch(b0 + kn + 128, 0, 1);
    __builtin_prefetch(b1 + kn + 128, 0, 1);
    V16H an, bn0, bn1;
    an.v    = *(const v16h*)(arow + kn + half * 16);
    bn0.v   = *(const v16h*)(b0 + kn + half * 16);
    bn1.v   = *(const v16h*)(b1 + kn + half * 16);

    acc0 = __builtin_amdgcn_wmma_f32_16x16x32_f16(false, a.v, false, bb0.v,
                                                  (short)0, acc0, false, false);
    acc1 = __builtin_amdgcn_wmma_f32_16x16x32_f16(false, a.v, false, bb1.v,
                                                  (short)0, acc1, false, false);
    a = an; bb0 = bn0; bb1 = bn1;
  }

  // epilogue: store both 16x16 tiles (optional residual add)
  #pragma unroll
  for (int sub = 0; sub < 2; ++sub) {
    int col = tileN + sub * 16 + nl;
    v8f accv = sub ? acc1 : acc0;
    if (col < Nstore) {
      #pragma unroll
      for (int r = 0; r < 8; ++r) {
        int row = tileM + half * 8 + r;
        size_t off = (size_t)row * ldc + col;
        float v = accv[r];
        if (Cin) v += Cin[off];
        Cout[off] = v;
      }
    }
  }
}

// ---------------------------------------------------------------------------
// 5) split qkv (f32 [M,3C]) -> Qb,Kb f16 [b,h,t,64] and Vt f16 [b,h,64,TP]
// ---------------------------------------------------------------------------
__global__ void splitqkv_kernel(const float* __restrict__ qkv,
                                f16* __restrict__ Qb, f16* __restrict__ Kb,
                                f16* __restrict__ Vt) {
  size_t i = (size_t)blockIdx.x * blockDim.x + threadIdx.x;
  if (i >= (size_t)GPT_M * GPT_C) return;
  int m = (int)(i / GPT_C);
  int c = (int)(i % GPT_C);
  int b = m / GPT_T, t = m % GPT_T;
  int h = c / GPT_HD, d = c % GPT_HD;
  const float* row = qkv + (size_t)m * (3 * GPT_C);
  size_t qk = (((size_t)b * GPT_H + h) * GPT_T + t) * GPT_HD + d;
  Qb[qk] = (f16)row[c];
  Kb[qk] = (f16)row[GPT_C + c];
  Vt[(((size_t)b * GPT_H + h) * GPT_HD + d) * GPT_TP + t] = (f16)row[2 * GPT_C + c];
}

__global__ void vtpad_kernel(f16* __restrict__ Vt) {
  size_t i = (size_t)blockIdx.x * blockDim.x + threadIdx.x;
  size_t npad = (size_t)GPT_B * GPT_H * GPT_HD * (GPT_TP - GPT_T);
  if (i >= npad) return;
  size_t r = i / (GPT_TP - GPT_T);
  size_t t = GPT_T + i % (GPT_TP - GPT_T);
  Vt[r * GPT_TP + t] = (f16)0.0f;
}

// ---------------------------------------------------------------------------
// 6) causal flash attention, one wave per 16-query tile.
//    K (32x64) and V^T (64x32) tiles are double-buffered in LDS via
//    per-wave global_load_async_to_lds_b128; next tile streams while the
//    current S/softmax/PV chain executes. s_wait_asynccnt <= 16 (in-order
//    completion) guarantees the current buffer landed. No barriers needed.
// ---------------------------------------------------------------------------
__global__ __launch_bounds__(32) void attn_kernel(const f16* __restrict__ Qb,
                                                  const f16* __restrict__ Kb,
                                                  const f16* __restrict__ Vt,
                                                  f16* __restrict__ Yb) {
  int qt = blockIdx.x;       // 0..T/16-1
  int hh = blockIdx.y;       // head
  int bb = blockIdx.z;       // batch
  int lane = threadIdx.x & 31;
  int half = lane >> 4;
  int nl   = lane & 15;

  size_t base  = ((size_t)bb * GPT_H + hh) * GPT_T * GPT_HD;
  size_t vbase = ((size_t)bb * GPT_H + hh) * GPT_HD * GPT_TP;
  int qbase = qt * 16;

  __shared__ __align__(16) f16 sK[2][32 * GPT_HD]; // [buf][key][d]
  __shared__ __align__(16) f16 sV[2][GPT_HD * 32]; // [buf][d][t_local]
  __shared__ __align__(16) f16 sP[16 * 32];

  // async staging of one 32-key block: 8+8 b128 chunks per lane
  auto stageK = [&](int buf, int k0) {
    #pragma unroll
    for (int i = 0; i < 8; ++i) {
      int c = i * 32 + lane;                 // chunk 0..255 (16B each)
      int key = k0 + (c >> 3);
      int keyc = key < (GPT_T - 1) ? key : (GPT_T - 1);
      int doff = (c & 7) * 8;
      uint64_t g = (uint64_t)(size_t)(Kb + base + (size_t)keyc * GPT_HD + doff);
      uint32_t d = (uint32_t)(size_t)&sK[buf][(c >> 3) * GPT_HD + doff];
      asm volatile("global_load_async_to_lds_b128 %0, %1, off"
                   :: "v"(d), "v"(g) : "memory");
    }
  };
  auto stageV = [&](int buf, int k0) {
    #pragma unroll
    for (int i = 0; i < 8; ++i) {
      int c = i * 32 + lane;                 // chunk 0..255 (16B each)
      int dd = c >> 2;
      int toff = (c & 3) * 8;
      uint64_t g = (uint64_t)(size_t)(Vt + vbase + (size_t)dd * GPT_TP + k0 + toff);
      uint32_t d = (uint32_t)(size_t)&sV[buf][dd * 32 + toff];
      asm volatile("global_load_async_to_lds_b128 %0, %1, off"
                   :: "v"(d), "v"(g) : "memory");
    }
  };

  // Q fragments (two 16x32 K-slices covering hd=64)
  const f16* qrow = Qb + base + (size_t)(qbase + nl) * GPT_HD;
  V16H qf[2];
  #pragma unroll
  for (int ks = 0; ks < 2; ++ks) {
    qf[ks].v = *(const v16h*)(qrow + ks * 32 + half * 16);
  }

  float m[8], l[8];
  v8f acc[4] = {};
  #pragma unroll
  for (int r = 0; r < 8; ++r) { m[r] = -3.0e38f; l[r] = 0.0f; }

  int nkb = (qbase + 16 + 31) >> 5;    // number of 32-key blocks (causal)
  stageK(0, 0);
  stageV(0, 0);
  int buf = 0;
  for (int kb = 0; kb < nkb; ++kb) {
    int k0 = kb * 32;
    int k0n = (kb + 1 < nkb) ? k0 + 32 : k0;   // last iter: redundant restage
    stageK(buf ^ 1, k0n);
    stageV(buf ^ 1, k0n);
    // wait until <=16 outstanding: the 16 chunks of the current buffer are
    // done (async loads complete in order), next 16 still in flight.
    asm volatile("s_wait_asynccnt 0x10" ::: "memory");

    // S = Q * K^T from LDS
    v8f S[2];
    #pragma unroll
    for (int sub = 0; sub < 2; ++sub) {
      const f16* krow = &sK[buf][(sub * 16 + nl) * GPT_HD];
      v8f s = {};
      #pragma unroll
      for (int ds = 0; ds < 2; ++ds) {
        v16h bf = *(const v16h*)(krow + ds * 32 + half * 16);
        s = __builtin_amdgcn_wmma_f32_16x16x32_f16(false, qf[ds].v, false, bf,
                                                   (short)0, s, false, false);
      }
      S[sub] = s;
    }
    // scale + causal mask
    #pragma unroll
    for (int sub = 0; sub < 2; ++sub) {
      int col = k0 + sub * 16 + nl;
      #pragma unroll
      for (int r = 0; r < 8; ++r) {
        int row = qbase + half * 8 + r;
        float v = S[sub][r] * 0.125f;   // 1/sqrt(64)
        S[sub][r] = (col <= row) ? v : -3.0e38f;
      }
    }
    // online softmax per row (16 lanes of a half share a row set)
    #pragma unroll
    for (int r = 0; r < 8; ++r) {
      float mx = fmaxf(S[0][r], S[1][r]);
      #pragma unroll
      for (int off = 8; off; off >>= 1) mx = fmaxf(mx, __shfl_xor(mx, off, 32));
      float mn = fmaxf(m[r], mx);
      float alpha = __expf(m[r] - mn);
      float p0 = __expf(S[0][r] - mn);
      float p1 = __expf(S[1][r] - mn);
      float rs = p0 + p1;
      #pragma unroll
      for (int off = 8; off; off >>= 1) rs += __shfl_xor(rs, off, 32);
      l[r] = l[r] * alpha + rs;
      m[r] = mn;
      #pragma unroll
      for (int nt = 0; nt < 4; ++nt) acc[nt][r] *= alpha;
      S[0][r] = p0; S[1][r] = p1;
    }
    // stage P (16x32) to LDS, reload in A-fragment layout
    #pragma unroll
    for (int sub = 0; sub < 2; ++sub)
      #pragma unroll
      for (int r = 0; r < 8; ++r)
        sP[(half * 8 + r) * 32 + sub * 16 + nl] = (f16)S[sub][r];
    __syncthreads();
    V16H pf;
    pf.v = *(const v16h*)(&sP[nl * 32 + half * 16]);
    // acc += P (16x32) @ V (32x64), 4 d-tiles of 16, V from LDS
    #pragma unroll
    for (int nt = 0; nt < 4; ++nt) {
      const f16* vrow = &sV[buf][(nt * 16 + nl) * 32 + half * 16];
      v16h bf = *(const v16h*)vrow;
      acc[nt] = __builtin_amdgcn_wmma_f32_16x16x32_f16(false, pf.v, false, bf,
                                                       (short)0, acc[nt], false, false);
    }
    __syncthreads();
    buf ^= 1;
  }

  // write normalized output as f16 [M, C] (heads interleaved back)
  #pragma unroll
  for (int nt = 0; nt < 4; ++nt)
    #pragma unroll
    for (int r = 0; r < 8; ++r) {
      int row = qbase + half * 8 + r;
      int c = hh * GPT_HD + nt * 16 + nl;
      Yb[((size_t)bb * GPT_T + row) * GPT_C + c] = (f16)(acc[nt][r] / l[r]);
    }
}

// ---------------------------------------------------------------------------
// 7) exact GELU, f32 in -> f16 out
// ---------------------------------------------------------------------------
__global__ void gelu_kernel(const float* __restrict__ in, f16* __restrict__ out,
                            size_t n) {
  size_t i = (size_t)blockIdx.x * blockDim.x + threadIdx.x;
  if (i >= n) return;
  float v = in[i];
  out[i] = (f16)(0.5f * v * (1.0f + erff(v * 0.70710678118654752f)));
}

// ---------------------------------------------------------------------------
// 8) per-row cross-entropy:  rowloss = logsumexp(logits_row) - logits[target]
// ---------------------------------------------------------------------------
__global__ __launch_bounds__(256) void rowloss_kernel(const float* __restrict__ logits,
                                                      const int* __restrict__ targets,
                                                      float* __restrict__ rowloss) {
  __shared__ float sm[8];
  int row = blockIdx.x;
  const float* lr = logits + (size_t)row * GPT_V;
  float mx = -3.0e38f;
  for (int c = threadIdx.x; c < GPT_V; c += 256) mx = fmaxf(mx, lr[c]);
  mx = blockReduce256(mx, sm, true);
  float se = 0.f;
  for (int c = threadIdx.x; c < GPT_V; c += 256) se += __expf(lr[c] - mx);
  se = blockReduce256(se, sm, false);
  if (threadIdx.x == 0)
    rowloss[row] = (mx + __logf(se)) - lr[targets[row]];
}

__global__ __launch_bounds__(256) void lossred_kernel(const float* __restrict__ rowloss,
                                                      float* __restrict__ out) {
  __shared__ float sm[8];
  float s = 0.f;
  for (int i = threadIdx.x; i < GPT_M; i += 256) s += rowloss[i];
  s = blockReduce256(s, sm, false);
  if (threadIdx.x == 0) out[0] = s * (1.0f / GPT_M);
}

// ---------------------------------------------------------------------------
// host orchestration
// ---------------------------------------------------------------------------
extern "C" void kernel_launch(void* const* d_in, const int* in_sizes, int n_in,
                              void* d_out, int out_size, void* d_ws, size_t ws_size,
                              hipStream_t stream) {
  const int*   idx         = (const int*)  d_in[0];
  const int*   targets     = (const int*)  d_in[1];
  const float* wte         = (const float*)d_in[2];
  const float* wpe         = (const float*)d_in[3];
  const float* ln1_w       = (const float*)d_in[4];
  const float* attn_w      = (const float*)d_in[5];
  const float* attn_proj_w = (const float*)d_in[6];
  const float* ln2_w       = (const float*)d_in[7];
  const float* fc_w        = (const float*)d_in[8];
  const float* mlp_proj_w  = (const float*)d_in[9];
  const float* lnf_w       = (const float*)d_in[10];
  const float* lm_head_w   = (const float*)d_in[11];

  float* logits = (float*)d_out;                       // [M, V]
  float* loss   = logits + (size_t)GPT_M * GPT_V;      // scalar at end

  // --- workspace carve-up (256B aligned regions) ---
  char* wsp = (char*)d_ws;
  auto carve = [&](size_t bytes) {
    char* p = wsp;
    wsp += (bytes + 255) & ~(size_t)255;
    return p;
  };
  float* x      = (float*)carve((size_t)GPT_M * GPT_C * 4);          // residual stream
  f16*   hb     = (f16*)  carve((size_t)GPT_M * GPT_C * 2);          // LN output (f16)
  float* qkv    = (float*)carve((size_t)GPT_M * 3 * GPT_C * 4);
  f16*   Qb     = (f16*)  carve((size_t)GPT_B * GPT_H * GPT_T * GPT_HD * 2);
  f16*   Kb     = (f16*)  carve((size_t)GPT_B * GPT_H * GPT_T * GPT_HD * 2);
  f16*   Vt     = (f16*)  carve((size_t)GPT_B * GPT_H * GPT_HD * GPT_TP * 2);
  f16*   Yb     = (f16*)  carve((size_t)GPT_M * GPT_C * 2);
  float* fcout  = (float*)carve((size_t)GPT_M * 4 * GPT_C * 4);
  f16*   fcact  = (f16*)  carve((size_t)GPT_M * 4 * GPT_C * 2);
  f16*   Wt     = (f16*)  carve((size_t)GPT_VPAD * GPT_C * 2);       // largest Wt (lm_head)
  float* rloss  = (float*)carve((size_t)GPT_M * 4);

  const int MC = GPT_M * GPT_C;

  // embeddings
  embed_kernel<<<(MC + 255) / 256, 256, 0, stream>>>(idx, wte, wpe, x);

  auto convw = [&](const float* W, int K, int N, int Npad) {
    size_t n = (size_t)Npad * K;
    convw_kernel<<<(unsigned)((n + 255) / 256), 256, 0, stream>>>(W, Wt, K, N, Npad);
  };
  auto gemm = [&](const f16* A, const float* Cin, float* Cout,
                  int M, int Npad, int K, int Nstore, int ldc) {
    dim3 g(Npad / 128, M / 32);
    gemm_f16_kernel<<<g, 256, 0, stream>>>(A, Wt, Cin, Cout, M, Npad, K, Nstore, ldc);
  };

  for (int l = 0; l < GPT_L; ++l) {
    // --- attention ---
    ln_f16_kernel<<<GPT_M, 256, 0, stream>>>(x, ln1_w + (size_t)l * GPT_C, hb);
    convw(attn_w + (size_t)l * GPT_C * 3 * GPT_C, GPT_C, 3 * GPT_C, 3 * GPT_C);
    gemm(hb, nullptr, qkv, GPT_M, 3 * GPT_C, GPT_C, 3 * GPT_C, 3 * GPT_C);

    splitqkv_kernel<<<(MC + 255) / 256, 256, 0, stream>>>(qkv, Qb, Kb, Vt);
    {
      size_t npad = (size_t)GPT_B * GPT_H * GPT_HD * (GPT_TP - GPT_T);
      vtpad_kernel<<<(unsigned)((npad + 255) / 256), 256, 0, stream>>>(Vt);
    }
    {
      dim3 g(GPT_T / 16, GPT_H, GPT_B);
      attn_kernel<<<g, 32, 0, stream>>>(Qb, Kb, Vt, Yb);
    }
    convw(attn_proj_w + (size_t)l * GPT_C * GPT_C, GPT_C, GPT_C, GPT_C);
    gemm(Yb, x, x, GPT_M, GPT_C, GPT_C, GPT_C, GPT_C);   // residual add

    // --- MLP ---
    ln_f16_kernel<<<GPT_M, 256, 0, stream>>>(x, ln2_w + (size_t)l * GPT_C, hb);
    convw(fc_w + (size_t)l * GPT_C * 4 * GPT_C, GPT_C, 4 * GPT_C, 4 * GPT_C);
    gemm(hb, nullptr, fcout, GPT_M, 4 * GPT_C, GPT_C, 4 * GPT_C, 4 * GPT_C);
    {
      size_t n = (size_t)GPT_M * 4 * GPT_C;
      gelu_kernel<<<(unsigned)((n + 255) / 256), 256, 0, stream>>>(fcout, fcact, n);
    }
    convw(mlp_proj_w + (size_t)l * 4 * GPT_C * GPT_C, 4 * GPT_C, GPT_C, GPT_C);
    gemm(fcact, x, x, GPT_M, GPT_C, 4 * GPT_C, GPT_C, GPT_C);  // residual add
  }

  // final LN + lm_head -> logits in d_out
  ln_f16_kernel<<<GPT_M, 256, 0, stream>>>(x, lnf_w, hb);
  convw(lm_head_w, GPT_C, GPT_V, GPT_VPAD);
  gemm(hb, nullptr, logits, GPT_M, GPT_VPAD, GPT_C, GPT_V, GPT_V);

  // cross-entropy loss
  rowloss_kernel<<<GPT_M, 256, 0, stream>>>(logits, targets, rloss);
  lossred_kernel<<<1, 256, 0, stream>>>(rloss, loss);
}
